// MWFBeamformer_24721831756490
// MI455X (gfx1250) — matrix operand order repaired and checked by reading, not compiled, not run on hardware
//
#include <hip/hip_runtime.h>
#include <cstdint>
#include <cstddef>

// ---------------- problem constants ----------------
#define N_FFT   512
#define HOP     128
#define PADW    256          // N_FFT/2 reflect pad
#define T_LEN   64000
#define BATCH   8
#define CHAN    8
#define FP      256          // freq bins kept (1..256)
#define NFR     501          // 1 + (64512-512)/128
#define HID     512
#define M_ROWS  (BATCH*NFR)  // 4008
#define M_PAD   4016         // next multiple of 16
#define OUTLEN  (N_FFT + HOP*(NFR-1))   // 64512
#define PI2F    6.28318530717958647692f

typedef _Float16 half_t;
typedef __attribute__((ext_vector_type(16))) _Float16 v16h;
typedef __attribute__((ext_vector_type(8)))  float    v8f;
typedef __attribute__((address_space(3))) _Float16 lds_half_t;

// ---------------- small helpers ----------------
__device__ __forceinline__ float hannw(int k) {
    return 0.5f - 0.5f * __cosf(PI2F * (float)k / (float)N_FFT);
}

// CDNA5 async global->LDS copy (ASYNCcnt-tracked), portable via inline asm
__device__ __forceinline__ void async_copy_b128(const void* g, uint32_t lds_off) {
    uint64_t ga = (uint64_t)(uintptr_t)g;
    asm volatile("global_load_async_to_lds_b128 %0, %1, off"
                 :: "v"(lds_off), "v"(ga)
                 : "memory");
}
__device__ __forceinline__ void wait_async0() {
    asm volatile("s_wait_asynccnt 0" ::: "memory");
}

// ---------------- kernel 0: zero overlap-add buffer, build wsum ----------------
__global__ void mwf_prep(float* __restrict__ yacc, float* __restrict__ wsum) {
    int idx = blockIdx.x * blockDim.x + threadIdx.x;
    if (idx < BATCH * OUTLEN) yacc[idx] = 0.0f;
    if (idx < OUTLEN) {
        float s = 0.0f;
        int base = idx / HOP;
        int rem  = idx % HOP;
#pragma unroll
        for (int d = 0; d < 4; ++d) {
            int j = base - d;
            int k = rem + HOP * d;
            if (j >= 0 && j < NFR && k < N_FFT) {
                float w = hannw(k);
                s += w * w;
            }
        }
        wsum[idx] = s;
    }
}

// ---------------- kernel 1: pack weights (K x N row-major) into per-lane WMMA B layout ----
// Wp[(kt*(N/16)+nt)*32*16 + lane*16 + e] = W[(kt*32 + 16*(lane>>4) + e)*N + nt*16 + (lane&15)]
template<int K, int N>
__global__ void mwf_pack_w(const float* __restrict__ W, half_t* __restrict__ Wp) {
    int idx = blockIdx.x * blockDim.x + threadIdx.x;
    if (idx >= K * N) return;
    int e    = idx & 15;
    int lane = (idx >> 4) & 31;
    int tile = idx >> 9;            // kt*(N/16) + nt
    int nt   = tile % (N / 16);
    int kt   = tile / (N / 16);
    int col  = nt * 16 + (lane & 15);
    int k    = kt * 32 + 16 * (lane >> 4) + e;
    Wp[idx]  = (half_t)W[(size_t)k * N + col];
}

// ---------------- kernel 2: STFT (windowed frame -> naive rDFT, bins 1..256) ----------------
__global__ void mwf_stft(const float* __restrict__ x, float2* __restrict__ X) {
    const int bid   = blockIdx.x;
    const int frame = bid % NFR;
    const int bc    = bid / NFR;
    const int c     = bc % CHAN;
    const int b     = bc / CHAN;

    __shared__ float fr[N_FFT];
    for (int k = threadIdx.x; k < N_FFT; k += blockDim.x) {
        int p = frame * HOP + k;
        int j = p - PADW;
        if (j < 0) j = -j;
        if (j >= T_LEN) j = 2*T_LEN - 2 - j;
        fr[k] = x[(size_t)(b*CHAN + c) * T_LEN + j] * hannw(k);
    }
    __syncthreads();

    const int fb = threadIdx.x;          // 0..255 -> bin fb+1
    const float ang = -PI2F * (float)(fb + 1) / (float)N_FFT;
    float sr = 0.0f, si = 0.0f;
    for (int k = 0; k < N_FFT; ++k) {
        float s, co;
        __sincosf(ang * (float)k, &s, &co);
        sr = fmaf(fr[k], co, sr);
        si = fmaf(fr[k], s,  si);
    }
    X[((size_t)(b*CHAN + c) * FP + fb) * NFR + frame] = make_float2(sr, si);
}

// ---------------- kernel 3: feature = |X_ref|, padded + f16 ----------------
__global__ void mwf_feat(const float2* __restrict__ X, half_t* __restrict__ featH) {
    int idx = blockIdx.x * blockDim.x + threadIdx.x;
    if (idx >= M_PAD * FP) return;
    int row = idx / FP, f = idx % FP;
    half_t v = (half_t)0.0f;
    if (row < M_ROWS) {
        int b = row / NFR, n = row % NFR;
        float2 z = X[((size_t)(b*CHAN + 0) * FP + f) * NFR + n];
        v = (half_t)sqrtf(z.x*z.x + z.y*z.y);
    }
    featH[idx] = v;
}

// ---------------- WMMA GEMM: 128 threads, 4 waves; block tile = 16(M) x 256(N) ----------
// A (M_PAD x KDIM, f16 row-major) staged to LDS via async-to-LDS; W pre-packed fragments.
// ACT=0: ReLU -> f16 out ; ACT=1: sigmoid -> f32 out
template<int KDIM, int NDIM, int ACT>
__global__ void __launch_bounds__(128)
mwf_gemm(const half_t* __restrict__ A, const half_t* __restrict__ Wp,
         const float* __restrict__ bias, half_t* __restrict__ outH,
         float* __restrict__ outF) {
    const int tid  = threadIdx.x;
    const int lane = tid & 31;
    const int wid  = tid >> 5;                      // 0..3
    const int m0   = blockIdx.y * 16;
    const int nbas = blockIdx.x * 256 + wid * 64;   // this wave's 64-col strip

    __shared__ half_t sA[16 * 32];                  // one 16x32 A tile (1 KB)

    v8f acc0 = {}, acc1 = {}, acc2 = {}, acc3 = {};
    const int row = lane & 15, hf = lane >> 4;

    for (int k0 = 0; k0 < KDIM; k0 += 32) {
        // ---- stage A tile: 16 rows x 64B, 64 async B128 copies (waves 0-1) ----
        if (tid < 64) {
            int r   = tid >> 2;                     // 0..15
            int seg = tid & 3;                      // 4 x 16B per row
            const half_t* g = A + (size_t)(m0 + r) * KDIM + k0 + seg * 8;
            uint32_t lds = (uint32_t)(uintptr_t)(lds_half_t*)&sA[r * 32 + seg * 8];
            async_copy_b128(g, lds);
        }
        wait_async0();
        __syncthreads();

        // ---- A fragment from LDS (16x32, contiguous K = 16*hf + e, same order as Wp) ----
        v16h a;
#pragma unroll
        for (int e = 0; e < 8; ++e) {
            a[e]     = sA[row * 32 + 16 * hf + e];       // K = 16*hf + e
            a[8 + e] = sA[row * 32 + 16 * hf + 8 + e];   // K = 16*hf + 8 + e
        }

        // ---- 4 pre-packed B fragments + 4 WMMA ----
        const size_t tbase = ((size_t)(k0 >> 5) * (NDIM / 16) + (nbas >> 4)) * 32 * 16
                           + (size_t)lane * 16;
        __builtin_prefetch(Wp + tbase + (size_t)(NDIM / 16) * 32 * 16, 0, 1);
        v16h b0 = *(const v16h*)(Wp + tbase);
        v16h b1 = *(const v16h*)(Wp + tbase + 512);      // +1 tile (32*16 halves)
        v16h b2 = *(const v16h*)(Wp + tbase + 1024);
        v16h b3 = *(const v16h*)(Wp + tbase + 1536);
        acc0 = __builtin_amdgcn_wmma_f32_16x16x32_f16(false, a, false, b0, (short)0, acc0, false, false);
        acc1 = __builtin_amdgcn_wmma_f32_16x16x32_f16(false, a, false, b1, (short)0, acc1, false, false);
        acc2 = __builtin_amdgcn_wmma_f32_16x16x32_f16(false, a, false, b2, (short)0, acc2, false, false);
        acc3 = __builtin_amdgcn_wmma_f32_16x16x32_f16(false, a, false, b3, (short)0, acc3, false, false);

        __syncthreads();    // protect sA before next iteration's overwrite
    }

    // ---- epilogue ----
    v8f accs[4] = {acc0, acc1, acc2, acc3};
#pragma unroll
    for (int t = 0; t < 4; ++t) {
        int col = nbas + t * 16 + (lane & 15);
        float bv = bias[col];
#pragma unroll
        for (int r = 0; r < 8; ++r) {
            int orow = m0 + r + 8 * hf;
            float v = accs[t][r] + bv;
            if (ACT == 0) outH[(size_t)orow * NDIM + col] = (half_t)fmaxf(v, 0.0f);
            else          outF[(size_t)orow * NDIM + col] = 1.0f / (1.0f + __expf(-v));
        }
    }
}

// ---------------- kernel 6: covariance accumulate + 8x8 complex solve ----------------
__global__ void __launch_bounds__(64)
mwf_cov_solve(const float2* __restrict__ X, const float* __restrict__ M,
              float2* __restrict__ bf) {
    const int b = blockIdx.x / FP;
    const int f = blockIdx.x % FP;
    const int tid = threadIdx.x;         // 64 threads: (c,d) pairs
    const int c = tid >> 3, d = tid & 7;

    const float2* Xc = X + ((size_t)(b*CHAN + c) * FP + f) * NFR;
    const float2* Xd = X + ((size_t)(b*CHAN + d) * FP + f) * NFR;
    const float*  Mr = M + (size_t)(b*NFR) * (2*FP) + f;

    float rtx = 0.f, rty = 0.f, rnx = 0.f, rny = 0.f;
    for (int n = 0; n < NFR; ++n) {
        float2 a  = Xc[n];
        float2 bb = Xd[n];
        float pr = a.x*bb.x + a.y*bb.y;     // a * conj(b)
        float pi = a.y*bb.x - a.x*bb.y;
        float mt = Mr[(size_t)n * (2*FP)];
        float mn = Mr[(size_t)n * (2*FP) + FP];
        rtx = fmaf(mt, pr, rtx); rty = fmaf(mt, pi, rty);
        rnx = fmaf(mn, pr, rnx); rny = fmaf(mn, pi, rny);
    }

    __shared__ float2 Aug[8][9];           // augmented [ Rt+Rn | rhs ]
    Aug[c][d] = make_float2(rtx + rnx, rty + rny);
    if (d == 0) Aug[c][8] = make_float2(rtx, rty);   // rhs = R_tgt[:, ref=0]
    __syncthreads();

    if (tid == 0) {
        for (int k = 0; k < 8; ++k) {
            int p = k;
            float best = Aug[k][k].x*Aug[k][k].x + Aug[k][k].y*Aug[k][k].y;
            for (int i = k+1; i < 8; ++i) {
                float m2 = Aug[i][k].x*Aug[i][k].x + Aug[i][k].y*Aug[i][k].y;
                if (m2 > best) { best = m2; p = i; }
            }
            if (p != k) {
                for (int j = k; j < 9; ++j) {
                    float2 t = Aug[k][j]; Aug[k][j] = Aug[p][j]; Aug[p][j] = t;
                }
            }
            float2 piv = Aug[k][k];
            float den = piv.x*piv.x + piv.y*piv.y + 1e-30f;
            float2 inv = make_float2(piv.x/den, -piv.y/den);
            for (int j = k; j < 9; ++j) {
                float2 z = Aug[k][j];
                Aug[k][j] = make_float2(z.x*inv.x - z.y*inv.y, z.x*inv.y + z.y*inv.x);
            }
            for (int i = 0; i < 8; ++i) {
                if (i == k) continue;
                float2 fac = Aug[i][k];
                for (int j = k; j < 9; ++j) {
                    float2 z = Aug[k][j];
                    Aug[i][j].x -= fac.x*z.x - fac.y*z.y;
                    Aug[i][j].y -= fac.x*z.y + fac.y*z.x;
                }
            }
        }
        for (int i = 0; i < 8; ++i)
            bf[((size_t)b * FP + f) * CHAN + i] = Aug[i][8];
    }
}

// ---------------- kernel 7: apply beamformer: Xe = sum_c X_c * conj(bf_c) ----------------
__global__ void mwf_apply(const float2* __restrict__ X, const float2* __restrict__ bf,
                          float2* __restrict__ Xe) {
    const int b = blockIdx.x / FP;
    const int f = blockIdx.x % FP;
    float2 w[CHAN];
#pragma unroll
    for (int c = 0; c < CHAN; ++c) w[c] = bf[((size_t)b * FP + f) * CHAN + c];
    for (int n = threadIdx.x; n < NFR; n += blockDim.x) {
        float er = 0.f, ei = 0.f;
#pragma unroll
        for (int c = 0; c < CHAN; ++c) {
            float2 xv = X[((size_t)(b*CHAN + c) * FP + f) * NFR + n];
            er += xv.x*w[c].x + xv.y*w[c].y;     // x * conj(w)
            ei += xv.y*w[c].x - xv.x*w[c].y;
        }
        Xe[((size_t)b * FP + f) * NFR + n] = make_float2(er, ei);
    }
}

// ---------------- kernel 8: iSTFT (inverse rDFT, DC=0) + windowed overlap-add ----------------
__global__ void __launch_bounds__(128)
mwf_istft(const float2* __restrict__ Xe, float* __restrict__ yacc) {
    const int b     = blockIdx.x / NFR;
    const int frame = blockIdx.x % NFR;
    __shared__ float2 sx[FP];
    for (int i = threadIdx.x; i < FP; i += blockDim.x)
        sx[i] = Xe[((size_t)b * FP + i) * NFR + frame];
    __syncthreads();

#pragma unroll
    for (int q = 0; q < 4; ++q) {
        int k = threadIdx.x + 128 * q;
        float sum = 0.0f;
        const float base = PI2F * (float)k / (float)N_FFT;
        for (int fb = 0; fb < FP-1; ++fb) {      // bins 1..255 (conjugate pairs)
            float s, co;
            __sincosf(base * (float)(fb + 1), &s, &co);
            sum += 2.0f * (sx[fb].x * co - sx[fb].y * s);
        }
        sum += sx[FP-1].x * ((k & 1) ? -1.0f : 1.0f);   // Nyquist bin 256
        float val = sum * (1.0f / (float)N_FFT) * hannw(k);
        atomicAdd(&yacc[(size_t)b * OUTLEN + frame * HOP + k], val);
    }
}

// ---------------- kernel 9: normalize + slice ----------------
__global__ void mwf_final(const float* __restrict__ yacc, const float* __restrict__ wsum,
                          float* __restrict__ out) {
    int idx = blockIdx.x * blockDim.x + threadIdx.x;
    if (idx >= BATCH * T_LEN) return;
    int b = idx / T_LEN, t = idx % T_LEN;
    int p = t + PADW;
    out[idx] = yacc[(size_t)b * OUTLEN + p] / fmaxf(wsum[p], 1e-10f);
}

// ---------------- host launch ----------------
extern "C" void kernel_launch(void* const* d_in, const int* in_sizes, int n_in,
                              void* d_out, int out_size, void* d_ws, size_t ws_size,
                              hipStream_t stream) {
    const float* x  = (const float*)d_in[0];
    const float* w1 = (const float*)d_in[1];
    const float* b1 = (const float*)d_in[2];
    const float* w2 = (const float*)d_in[3];
    const float* b2 = (const float*)d_in[4];
    float* out = (float*)d_out;

    char* ws = (char*)d_ws;
    size_t off = 0;
    auto carve = [&](size_t bytes) -> void* {
        void* p = ws + off;
        off = (off + bytes + 255) & ~(size_t)255;
        return p;
    };

    float2* X     = (float2*)carve((size_t)BATCH*CHAN*FP*NFR * sizeof(float2));   // ~63 MB
    float2* Xe    = (float2*)carve((size_t)BATCH*FP*NFR * sizeof(float2));
    half_t* featH = (half_t*)carve((size_t)M_PAD*FP * sizeof(half_t));
    half_t* Hh    = (half_t*)carve((size_t)M_PAD*HID * sizeof(half_t));
    float*  Mm    = (float*) carve((size_t)M_PAD*(2*FP) * sizeof(float));
    half_t* w1p   = (half_t*)carve((size_t)FP*HID * sizeof(half_t));
    half_t* w2p   = (half_t*)carve((size_t)HID*(2*FP) * sizeof(half_t));
    float2* bf    = (float2*)carve((size_t)BATCH*FP*CHAN * sizeof(float2));
    float*  yacc  = (float*) carve((size_t)BATCH*OUTLEN * sizeof(float));
    float*  wsum  = (float*) carve((size_t)OUTLEN * sizeof(float));
    (void)ws_size; (void)in_sizes; (void)n_in; (void)out_size;

    // 0) zero overlap-add buffer + window-sum
    mwf_prep<<<(BATCH*OUTLEN + 255)/256, 256, 0, stream>>>(yacc, wsum);
    // 1) weights -> packed f16 WMMA-fragment layout
    mwf_pack_w<FP,  HID><<<(FP*HID   + 255)/256, 256, 0, stream>>>(w1, w1p);
    mwf_pack_w<HID, 2*FP><<<(HID*2*FP + 255)/256, 256, 0, stream>>>(w2, w2p);
    // 2) STFT
    mwf_stft<<<BATCH*CHAN*NFR, 256, 0, stream>>>(x, X);
    // 3) features
    mwf_feat<<<(M_PAD*FP + 255)/256, 256, 0, stream>>>(X, featH);
    // 4) MLP layer 1 (WMMA, async-LDS staged A, ReLU -> f16)
    mwf_gemm<FP,  HID, 0><<<dim3(HID/256,  M_PAD/16), 128, 0, stream>>>(featH, w1p, b1, Hh, nullptr);
    // 5) MLP layer 2 (WMMA, sigmoid -> f32 masks)
    mwf_gemm<HID, 2*FP, 1><<<dim3((2*FP)/256, M_PAD/16), 128, 0, stream>>>(Hh, w2p, b2, nullptr, Mm);
    // 6) covariance + solve
    mwf_cov_solve<<<BATCH*FP, 64, 0, stream>>>(X, Mm, bf);
    // 7) apply beamformer
    mwf_apply<<<BATCH*FP, 128, 0, stream>>>(X, bf, Xe);
    // 8) iSTFT + overlap-add
    mwf_istft<<<BATCH*NFR, 128, 0, stream>>>(Xe, yacc);
    // 9) normalize + slice
    mwf_final<<<(BATCH*T_LEN + 255)/256, 256, 0, stream>>>(yacc, wsum, out);
}
